// FixedPointIterationGapN_88175678587251
// MI455X (gfx1250) — matrix operand order repaired and checked by reading, not compiled, not run on hardware
//
#include <hip/hip_runtime.h>
#include <stdint.h>

typedef __attribute__((ext_vector_type(16))) _Float16 v16h;
typedef __attribute__((ext_vector_type(8)))  float    v8f;
typedef __attribute__((ext_vector_type(4)))  int      v4i;

#define TS      4
#define LOG2E   1.44269504088896340736f
#define LN2     0.69314718055994530942f
// unified staging slot layout: dword = SIDX(i,j) + tsl ; bank-conflict free for
// fwd (fixed i, lanes over j) and bwd (lanes over i, fixed j) fragment gathers.
#define SIDX(i,j) (((i)*33 + (j) + (((j)&16)>>1))*4)
#define STG_DW    4252   // (31*33+31+8+1)*4

#if __has_builtin(__builtin_amdgcn_global_load_async_to_lds_b32)
#define HAVE_ASYNC 1
#else
#define HAVE_ASYNC 0
#endif
#if HAVE_ASYNC && __has_builtin(__builtin_amdgcn_global_load_async_to_lds_b128)
#define HAVE_A128 1
#else
#define HAVE_A128 0
#endif

typedef __attribute__((address_space(1))) int gint_t;   // global addrspace
typedef __attribute__((address_space(3))) int lint_t;   // LDS addrspace
typedef __attribute__((address_space(1))) v4i gv4i_t;   // global int4
typedef __attribute__((address_space(3))) v4i lv4i_t;   // LDS int4

#if HAVE_ASYNC
#if __has_builtin(__builtin_amdgcn_s_wait_asynccnt)
#define ASY_WAIT0  do { __builtin_amdgcn_s_wait_asynccnt(0);  asm volatile("" ::: "memory"); } while (0)
#define ASY_WAIT32 do { __builtin_amdgcn_s_wait_asynccnt(32); asm volatile("" ::: "memory"); } while (0)
#else
#define ASY_WAIT0  asm volatile("s_wait_asynccnt 0"  ::: "memory")
#define ASY_WAIT32 asm volatile("s_wait_asynccnt 32" ::: "memory")
#endif
#else
#define ASY_WAIT0  do {} while (0)
#define ASY_WAIT32 do {} while (0)
#endif

struct Job {
  const float* u;
  const float* pw;
  float* out_acc;
  float* m0;                    // non-null only for level 0
  int L, T;                     // chains per batch, chain length
  int uB, uC, uT, uCh;          // unary strides (elements)
  int pB, pI, pJ, pT, pCh;      // pairwise strides (elements); note pI == 32*pJ
  int si, sj, stride, vert;     // scatter placement into the 128x128 fine grid
};
struct JobPack { Job j[8]; };

__global__ void __launch_bounds__(64)
chain_kernel(JobPack jp, const float* __restrict__ gp) {
  __shared__ _Float16 fb[128 * 32];                    // forward messages (f16)
  __shared__ _Float16 bbuf[128 * 32];                  // backward messages (f16)
  __shared__ __align__(16) float stg[2][2][STG_DW];    // [wave][buf] staging tiles
  __shared__ __align__(64) _Float16 atw[2][32];        // per-wave a_i broadcast table

  const Job J    = jp.j[blockIdx.y];
  const int lane = threadIdx.x & 31;
  const int wave = threadIdx.x >> 5;
  const int b    = blockIdx.x / J.L;
  const int l    = blockIdx.x % J.L;

  float g = gp[0]; g = (g < 0.01f) ? 0.01f : g;        // jnp.clip(gamma, 0.01)
  const float ig  = 1.0f / g;
  const float igl = ig * LOG2E;

  const float* ub = J.u  + (size_t)b * J.uB + (size_t)l * J.uCh + (size_t)lane * J.uC;
  const float* pb = J.pw + (size_t)b * J.pB + (size_t)l * J.pCh;

  const int T = J.T, nst = T - 1;
  const int ntiles = (nst + TS - 1) / TS;
  const bool fast = HAVE_A128 && (J.pT == 1);          // contiguous-t tensors
  auto t0of = [&](int tt) { int t0 = tt * TS, mx = nst - TS; return t0 > mx ? mx : t0; };

  // ---- b128 staging: 32 instrs/tile, 16B coalesced, double-bufferable ----
  auto issue128 = [&](int t0c, float* dst) {
#if HAVE_A128
    const float* gl = pb + (size_t)t0c + (size_t)lane * J.pJ;          // pT==1
    float* ld = dst + (size_t)(lane + ((lane & 16) >> 1)) * 4;
#pragma unroll 8
    for (int i = 0; i < 32; ++i)
      __builtin_amdgcn_global_load_async_to_lds_b128(
          (gv4i_t*)(float*)(gl + (size_t)i * J.pI), (lv4i_t*)(ld + i * 33 * 4), 0, 0);
#endif
  };
  // ---- b32 gather staging (any strides), same LDS layout ----
  auto issue32 = [&](int t0c, float* dst) {
    const int jb = lane >> 2, t4 = lane & 3;
    const float* gl = pb + (size_t)(t0c + t4) * J.pT + (size_t)jb * J.pJ;
    float* ld = dst + lane;
    const size_t pj8 = (size_t)J.pJ * 8;               // pI == 32*pJ -> offset = q*8*pJ
#pragma unroll 8
    for (int q = 0; q < 128; ++q) {
      const int i = q >> 2, rem = q & 3;
      const int U = (i * 33 + rem * 8 + ((rem >> 1) << 3)) * 4;
#if HAVE_ASYNC
      __builtin_amdgcn_global_load_async_to_lds_b32(
          (gint_t*)(float*)(gl + (size_t)q * pj8), (lint_t*)(ld + U), 0, 0);
#else
      ld[U] = gl[(size_t)q * pj8];
#endif
    }
  };

  // ---- one smoothed-LSE step: returns m + g*ln( sum_i a_i * exp(pw/g) ) ----
  auto lse_step = [&](const float* st, float vin, int tsl, bool tr) -> float {
    float m = vin;
#pragma unroll
    for (int s = 16; s; s >>= 1) m = fmaxf(m, __shfl_xor(m, s, 32));
    const float a = __builtin_amdgcn_exp2f((vin - m) * igl);
    atw[wave][lane] = (_Float16)a;
    asm volatile("" ::: "memory");
    const uint32_t* at = (const uint32_t*)&atw[wave][0];
    const int ab = (lane < 16) ? 0 : 8;
    union { uint32_t w[8]; v16h h; } A;
#pragma unroll
    for (int v = 0; v < 4; ++v) { A.w[v] = at[ab + v]; A.w[4 + v] = at[4 + ab + v]; }
    const int col = lane & 15;
    const int kb  = (lane < 16) ? 0 : 16;
    v16h B1, B2;
#pragma unroll
    for (int v = 0; v < 8; ++v) {
      const int k = kb + 2 * v;
      float x0, x1, y0, y1;
      if (!tr) {                                       // B = X (cols 0-15 / 16-31)
        x0 = st[SIDX(k,     col     ) + tsl];
        x1 = st[SIDX(k + 1, col     ) + tsl];
        y0 = st[SIDX(k,     col + 16) + tsl];
        y1 = st[SIDX(k + 1, col + 16) + tsl];
      } else {                                         // B = X^T
        x0 = st[SIDX(col,      k    ) + tsl];
        x1 = st[SIDX(col,      k + 1) + tsl];
        y0 = st[SIDX(col + 16, k    ) + tsl];
        y1 = st[SIDX(col + 16, k + 1) + tsl];
      }
      B1[2*v]   = (_Float16)__builtin_amdgcn_exp2f(x0 * igl);
      B1[2*v+1] = (_Float16)__builtin_amdgcn_exp2f(x1 * igl);
      B2[2*v]   = (_Float16)__builtin_amdgcn_exp2f(y0 * igl);
      B2[2*v+1] = (_Float16)__builtin_amdgcn_exp2f(y1 * igl);
    }
    v8f Z = {};
    v8f D1 = __builtin_amdgcn_wmma_f32_16x16x32_f16(false, A.h, false, B1, (short)0, Z, false, false);
    v8f D2 = __builtin_amdgcn_wmma_f32_16x16x32_f16(false, A.h, false, B2, (short)0, Z, false, false);
    const float s = (lane < 16) ? D1[0] : D2[0];
    return m + g * (LN2 * __builtin_amdgcn_logf(s));
  };

  if (wave == 0) {                                     // ---- forward ----
    float f = ub[0];
    fb[lane] = (_Float16)f;
    if (fast) issue128(t0of(0), stg[wave][0]);
    for (int tt = 0; tt < ntiles; ++tt) {
      const float* st;
      if (fast) {
        if (tt + 1 < ntiles) { issue128(t0of(tt + 1), stg[wave][(tt + 1) & 1]); ASY_WAIT32; }
        else                 { ASY_WAIT0; }
        st = stg[wave][tt & 1];
      } else {
        issue32(t0of(tt), stg[wave][0]); ASY_WAIT0; st = stg[wave][0];
      }
      const int t0c = t0of(tt);
      const int te  = ((tt + 1) * TS < nst) ? (tt + 1) * TS : nst;
      for (int t = tt * TS; t < te; ++t) {
        f = ub[(size_t)(t + 1) * J.uT] + lse_step(st, f, t - t0c, false);
        fb[(t + 1) * 32 + lane] = (_Float16)f;
      }
    }
  } else {                                             // ---- backward ----
    float bv = ub[(size_t)(T - 1) * J.uT];
    bbuf[(T - 1) * 32 + lane] = (_Float16)bv;
    if (fast) issue128(t0of(ntiles - 1), stg[wave][(ntiles - 1) & 1]);
    for (int tt = ntiles - 1; tt >= 0; --tt) {
      const float* st;
      if (fast) {
        if (tt - 1 >= 0) { issue128(t0of(tt - 1), stg[wave][(tt - 1) & 1]); ASY_WAIT32; }
        else             { ASY_WAIT0; }
        st = stg[wave][tt & 1];
      } else {
        issue32(t0of(tt), stg[wave][0]); ASY_WAIT0; st = stg[wave][0];
      }
      const int t0c = t0of(tt);
      const int te  = ((tt + 1) * TS < nst) ? (tt + 1) * TS : nst;
      for (int t = te - 1; t >= tt * TS; --t) {
        bv = ub[(size_t)t * J.uT] + lse_step(st, bv, t - t0c, true);
        bbuf[t * 32 + lane] = (_Float16)bv;
      }
    }
  }
  __syncthreads();

  // ---- marginals: softmax over C of (f+b-u)/g; write or scatter-add ----
  for (int t = wave; t < T; t += 2) {
    const float fv = (float)fb[t * 32 + lane];
    const float bv = (float)bbuf[t * 32 + lane];
    const float uv = ub[(size_t)t * J.uT];
    const float x  = (fv + bv - uv) * ig;
    float mx = x;
#pragma unroll
    for (int s = 16; s; s >>= 1) mx = fmaxf(mx, __shfl_xor(mx, s, 32));
    const float e = __builtin_amdgcn_exp2f((x - mx) * LOG2E);
    float ssum = e;
#pragma unroll
    for (int s = 16; s; s >>= 1) ssum += __shfl_xor(ssum, s, 32);
    const float p = e / ssum;
    int hh, ww;
    if (J.vert) { hh = J.si + J.stride * t; ww = J.sj + J.stride * l; }
    else        { hh = J.si + J.stride * l; ww = J.sj + J.stride * t; }
    const size_t oi = (((size_t)(b * 32 + lane)) * 128 + hh) * 128 + ww;
    if (J.m0) { J.m0[oi] = p; J.out_acc[oi] = p; }
    else      { J.out_acc[oi] += p; }
  }
}

// transpose last two dims: in [M][R][Wd] -> out [M][Wd][R]
__global__ void transpose_kernel(const float* __restrict__ in, float* __restrict__ outp,
                                 int R, int Wd) {
  __shared__ float tile[32][33];
  const size_t base = (size_t)blockIdx.z * R * Wd;
  const int x = blockIdx.x * 32 + threadIdx.x;
#pragma unroll
  for (int dy = 0; dy < 32; dy += 8) {
    const int r = blockIdx.y * 32 + dy + threadIdx.y;
    if (r < R && x < Wd) tile[dy + threadIdx.y][threadIdx.x] = in[base + (size_t)r * Wd + x];
  }
  __syncthreads();
  const int ro = blockIdx.y * 32 + threadIdx.x;
#pragma unroll
  for (int dy = 0; dy < 32; dy += 8) {
    const int w = blockIdx.x * 32 + dy + threadIdx.y;
    if (w < Wd && ro < R) outp[base + (size_t)w * R + ro] = tile[threadIdx.x][dy + threadIdx.y];
  }
}

__global__ void finalize_kernel(const float* __restrict__ uh0, const float* __restrict__ uv0,
                                float* __restrict__ out, int n) {
  const int i = blockIdx.x * blockDim.x + threadIdx.x;
  if (i >= n) return;
  float* o0 = out;
  float* o1 = out + (size_t)n;
  float* o2 = out + 2 * (size_t)n;
  float* o3 = out + 3 * (size_t)n;
  const float alpha = 1.0f / 128.0f;
  const float avg = 0.25f * (o2[i] + o3[i]);
  o0[i] = uh0[i] - alpha * (o0[i] - avg);
  o1[i] = uv0[i] - alpha * (o1[i] - avg);
}

static void make_jobs_level(Job* jh, Job* jv,
                            const float* uh, const float* uv,
                            const float* ph, const float* pv, const float* pvT,
                            float* acc_h, float* acc_v, float* m0h, float* m0v,
                            int H, int W, int si, int sj, int stride) {
  jh->u = uh; jh->pw = ph; jh->out_acc = acc_h; jh->m0 = m0h;
  jh->L = H; jh->T = W;
  jh->uB = 32 * H * W; jh->uC = H * W; jh->uT = 1; jh->uCh = W;
  jh->pB = 32 * 32 * H * (W - 1); jh->pI = 32 * H * (W - 1);
  jh->pJ = H * (W - 1); jh->pT = 1; jh->pCh = (W - 1);
  jh->si = si; jh->sj = sj; jh->stride = stride; jh->vert = 0;

  jv->u = uv; jv->out_acc = acc_v; jv->m0 = m0v;
  jv->L = W; jv->T = H;
  jv->uB = 32 * H * W; jv->uC = H * W; jv->uT = W; jv->uCh = 1;
  if (pvT) {            // pre-transposed: [B,C,C,W,H-1], contiguous t
    jv->pw = pvT;
    jv->pB = 32 * 32 * W * (H - 1); jv->pI = 32 * W * (H - 1);
    jv->pJ = W * (H - 1); jv->pT = 1; jv->pCh = (H - 1);
  } else {              // original: [B,C,C,H-1,W]
    jv->pw = pv;
    jv->pB = 32 * 32 * (H - 1) * W; jv->pI = 32 * (H - 1) * W;
    jv->pJ = (H - 1) * W; jv->pT = W; jv->pCh = 1;
  }
  jv->si = si; jv->sj = sj; jv->stride = stride; jv->vert = 1;
}

extern "C" void kernel_launch(void* const* d_in, const int* in_sizes, int n_in,
                              void* d_out, int out_size, void* d_ws, size_t ws_size,
                              hipStream_t stream) {
  (void)in_sizes; (void)n_in; (void)out_size;
  const int n = 2 * 32 * 128 * 128;
  float* out = (float*)d_out;
  float* o0 = out;
  float* o1 = out + (size_t)n;
  float* o2 = out + 2 * (size_t)n;
  float* o3 = out + 3 * (size_t)n;
  const float* gamma = (const float*)d_in[20];

  // optional pre-transpose of vertical pairwise tensors into d_ws (makes all
  // chains t-contiguous -> b128 async staging + double buffering)
  const size_t e0 = (size_t)2 * 32 * 32 * 127 * 128;   // level-0 pv elements
  const size_t ec = (size_t)2 * 32 * 32 * 63 * 64;     // coarse pv elements
  const bool dotr = ws_size >= (e0 + 4 * ec) * sizeof(float);
  float* pvT[5] = {nullptr, nullptr, nullptr, nullptr, nullptr};
  if (dotr) {
    float* w = (float*)d_ws;
    pvT[0] = w; w += e0;
    transpose_kernel<<<dim3(4, 4, 2048), dim3(32, 8), 0, stream>>>(
        (const float*)d_in[3], pvT[0], 127, 128);
    for (int lvl = 1; lvl <= 4; ++lvl) {
      pvT[lvl] = w; w += ec;
      transpose_kernel<<<dim3(2, 2, 2048), dim3(32, 8), 0, stream>>>(
          (const float*)d_in[4 * lvl + 3], pvT[lvl], 63, 64);
    }
  }

  JobPack p0{};
  make_jobs_level(&p0.j[0], &p0.j[1],
                  (const float*)d_in[0], (const float*)d_in[1],
                  (const float*)d_in[2], (const float*)d_in[3], pvT[0],
                  o2, o3, o0, o1, 128, 128, 0, 0, 1);
  chain_kernel<<<dim3(256, 2), 64, 0, stream>>>(p0, gamma);   // level 0 (init acc)

  JobPack pc{};
  const int off[4][2] = {{0, 0}, {0, 1}, {1, 0}, {1, 1}};
  for (int lvl = 1; lvl <= 4; ++lvl) {
    make_jobs_level(&pc.j[(lvl - 1) * 2], &pc.j[(lvl - 1) * 2 + 1],
                    (const float*)d_in[4 * lvl], (const float*)d_in[4 * lvl + 1],
                    (const float*)d_in[4 * lvl + 2], (const float*)d_in[4 * lvl + 3], pvT[lvl],
                    o2, o3, nullptr, nullptr, 64, 64,
                    off[lvl - 1][0], off[lvl - 1][1], 2);
  }
  chain_kernel<<<dim3(128, 8), 64, 0, stream>>>(pc, gamma);   // coarse, scatter-add

  finalize_kernel<<<(n + 255) / 256, 256, 0, stream>>>(
      (const float*)d_in[0], (const float*)d_in[1], out, n);
}
